// DCellNet_38551626449277
// MI455X (gfx1250) — matrix operand order repaired and checked by reading, not compiled, not run on hardware
//
#include <hip/hip_runtime.h>

typedef __bf16 bf16_t;
typedef __attribute__((ext_vector_type(16))) __bf16 v16bf;
typedef __attribute__((ext_vector_type(8)))  __bf16 v8bf;
typedef __attribute__((ext_vector_type(8)))  float  v8f;

// ---- problem constants ----
#define NB    2048          // batch
#define NFR   3000          // real feature dim
#define NFP   3008          // padded (mult of 32)
#define NL    384           // leaves
#define NG    10            // genes per leaf
#define NGP   16            // padded
#define N1    6144          // NL*NGP
#define NDL   15
#define NM    16
#define NCDL  360           // 24*15
#define KMID  384           // padded
#define NDM   72
#define NDMP  80            // padded to 16
#define KROOT 1152          // 16*72 (already mult of 32)
#define NDR   900
#define NDRP  912           // padded to 16

#if defined(__AMDGCN__) && __has_builtin(__builtin_amdgcn_global_load_async_to_lds_b128)
#define USE_ASYNC_LDS 1
#else
#define USE_ASYNC_LDS 0
#endif

#if USE_ASYNC_LDS
typedef int v4i_vs __attribute__((vector_size(16)));
typedef __attribute__((address_space(1))) v4i_vs* as1_v4i_p;   // global int4*
typedef __attribute__((address_space(3))) v4i_vs* as3_v4i_p;   // LDS int4*
#endif

__device__ __forceinline__ bf16_t bf16_zero() {
    return __builtin_bit_cast(bf16_t, (unsigned short)0);
}
__device__ __forceinline__ bf16_t f2bf(float f) {
    union { float f; unsigned u; } c; c.f = f;
    unsigned r = c.u + 0x7FFFu + ((c.u >> 16) & 1u);   // round-to-nearest-even
    return __builtin_bit_cast(bf16_t, (unsigned short)(r >> 16));
}

__device__ __forceinline__ void async_load_b128(const void* g, void* l) {
#if USE_ASYNC_LDS
    __builtin_amdgcn_global_load_async_to_lds_b128(
        (as1_v4i_p)(unsigned long long)(size_t)g,
        (as3_v4i_p)(unsigned int)(size_t)l,
        0, 0);
#endif
}
__device__ __forceinline__ void wait_async0() {
#if defined(__AMDGCN__)
#if __has_builtin(__builtin_amdgcn_s_wait_asynccnt)
    __builtin_amdgcn_s_wait_asynccnt(0);
#else
    asm volatile("s_wait_asynccnt 0x0" ::: "memory");
#endif
#endif
}

// ---------------- cast / gather kernels ----------------
__global__ void cast_x_kernel(const float* __restrict__ x, bf16_t* __restrict__ xb) {
    int gid = blockIdx.x * 256 + threadIdx.x;
    if (gid >= NB * NFP) return;
    int k = gid % NFP, b = gid / NFP;
    float v = (k < NFR) ? x[(size_t)b * NFR + k] : 0.f;
    xb[gid] = f2bf(v);
}

__global__ void cast_wd_kernel(const float* __restrict__ Wd, bf16_t* __restrict__ wdT) {
    int gid = blockIdx.x * 256 + threadIdx.x;
    if (gid >= N1 * NFP) return;
    int k = gid % NFP, n = gid / NFP;
    int l = n >> 4, g = n & 15;
    float v = (g < NG && k < NFR) ? Wd[((size_t)l * NFR + k) * NG + g] : 0.f;
    wdT[gid] = f2bf(v);
}

__global__ void build_biasd_kernel(const float* __restrict__ bd, float* __restrict__ biasd) {
    int n = blockIdx.x * 256 + threadIdx.x;
    if (n >= N1) return;
    int l = n >> 4, g = n & 15;
    biasd[n] = (g < NG) ? bd[l * NG + g] : 0.f;
}

__global__ void cast_cin_kernel(const float* __restrict__ hl, bf16_t* __restrict__ cinb) {
    int gid = blockIdx.x * 256 + threadIdx.x;
    if (gid >= NM * NB * KMID) return;
    int k = gid % KMID;
    int b = (gid / KMID) % NB;
    int m = gid / (KMID * NB);
    float v = 0.f;
    if (k < NCDL) {
        int c = k / NDL, dl = k % NDL;
        v = hl[((size_t)(m * 24 + c) * NB + b) * NDL + dl];
    }
    cinb[gid] = f2bf(v);
}

__global__ void cast_wm_kernel(const float* __restrict__ Wm, bf16_t* __restrict__ wmT) {
    int gid = blockIdx.x * 256 + threadIdx.x;
    if (gid >= NM * NDMP * KMID) return;
    int k = gid % KMID;
    int n = (gid / KMID) % NDMP;
    int m = gid / (KMID * NDMP);
    float v = (n < NDM && k < NCDL) ? Wm[((size_t)m * NCDL + k) * NDM + n] : 0.f;
    wmT[gid] = f2bf(v);
}

__global__ void cast_rin_kernel(const float* __restrict__ hm, bf16_t* __restrict__ rinb) {
    int gid = blockIdx.x * 256 + threadIdx.x;
    if (gid >= NB * KROOT) return;
    int k = gid % KROOT, b = gid / KROOT;
    int m = k / NDM, dm = k % NDM;
    rinb[gid] = f2bf(hm[((size_t)m * NB + b) * NDM + dm]);
}

__global__ void cast_wr_kernel(const float* __restrict__ Wr, bf16_t* __restrict__ wrT) {
    int gid = blockIdx.x * 256 + threadIdx.x;
    if (gid >= NDRP * KROOT) return;
    int k = gid % KROOT, n = gid / KROOT;
    float v = (n < NDR) ? Wr[(size_t)k * NDR + n] : 0.f;
    wrT[gid] = f2bf(v);
}

// ---------------- WMMA bf16 GEMM: C[M x Nreal] = A[M x K] * Bt[N x K]^T ----------------
// Tile 128x128, 256 threads (8 waves). Wave w owns M-subtile w (16 rows) and all
// 8 N-subtiles. A,Bt are bf16; Bt is N-major (row n holds K contiguous).
// Double-buffered software pipeline; async global->LDS when available.
__global__ __launch_bounds__(256) void wmma_gemm_kernel(
    const bf16_t* __restrict__ A, const bf16_t* __restrict__ Bt,
    float* __restrict__ C, const float* __restrict__ bias,
    int K, int Ntot, int Nreal, int ldC, int useTanh,
    long long sA, long long sB, long long sC, long long sBias)
{
    const int z = blockIdx.z;
    A += sA * z; Bt += sB * z; C += sC * z; bias += sBias * z;

    const int tid    = threadIdx.x;
    const int lid    = tid & 31;
    const int wave   = tid >> 5;
    const int half   = lid >> 4;    // which 16-lane half of the wave
    const int lane16 = lid & 15;

    const int m0 = blockIdx.x * 128;
    const int n0 = blockIdx.y * 128;

    // staging assignment: thread owns rows srow and srow+64, 16B column chunk scol
    const int srow = tid >> 2;
    const int scol = (tid & 3) << 3;           // bf16 elements (16 bytes)
    const size_t aOff0 = (size_t)(m0 + srow) * K + scol;
    const size_t aOff1 = (size_t)(m0 + srow + 64) * K + scol;
    const bool  bOk0  = (n0 + srow) < Ntot;
    const bool  bOk1  = (n0 + srow + 64) < Ntot;
    const size_t bOff0 = (size_t)(n0 + srow) * K + scol;
    const size_t bOff1 = (size_t)(n0 + srow + 64) * K + scol;
    const int lA0 = srow * 32 + scol;
    const int lA1 = (srow + 64) * 32 + scol;

    v8f acc[8];
    #pragma unroll
    for (int j = 0; j < 8; ++j)
        #pragma unroll
        for (int r = 0; r < 8; ++r) acc[j][r] = 0.f;

    const int kTiles = K >> 5;

    auto compute = [&](const bf16_t* lA, const bf16_t* lB) {
        // A fragment (16x32 bf16): lane half h, row m: v0..3 hold K[h*8..h*8+7],
        // v4..7 hold K[16+h*8 .. 16+h*8+7]
        const bf16_t* arow = &lA[(wave * 16 + lane16) * 32];
        v8bf alo = *(const v8bf*)(arow + half * 8);
        v8bf ahi = *(const v8bf*)(arow + 16 + half * 8);
        v16bf afrag = __builtin_shufflevector(alo, ahi,
            0, 1, 2, 3, 4, 5, 6, 7, 8, 9, 10, 11, 12, 13, 14, 15);
        #pragma unroll
        for (int j = 0; j < 8; ++j) {
            // B fragment (32x16 bf16): lane n=lane16, v0..3 hold K[h*8..h*8+7], v4..7 hold K[16+h*8..16+h*8+7]
            const bf16_t* brow = &lB[(j * 16 + lane16) * 32];
            v8bf blo = *(const v8bf*)(brow + half * 8);
            v8bf bhi = *(const v8bf*)(brow + 16 + half * 8);
            v16bf bfrag = __builtin_shufflevector(blo, bhi,
                0, 1, 2, 3, 4, 5, 6, 7, 8, 9, 10, 11, 12, 13, 14, 15);
            acc[j] = __builtin_amdgcn_wmma_f32_16x16x32_bf16(
                false, afrag, false, bfrag, (short)0, acc[j], false, false);
        }
    };

#if USE_ASYNC_LDS
    __shared__ __align__(32) bf16_t ldsA[2][128 * 32];
    __shared__ __align__(32) bf16_t ldsB[2][128 * 32];

    // zero guard rows (padded N) of both B buffers once; never async-written after
    {
        v8bf vz;
        #pragma unroll
        for (int i = 0; i < 8; ++i) vz[i] = bf16_zero();
        if (!bOk0) { *(v8bf*)&ldsB[0][lA0] = vz; *(v8bf*)&ldsB[1][lA0] = vz; }
        if (!bOk1) { *(v8bf*)&ldsB[0][lA1] = vz; *(v8bf*)&ldsB[1][lA1] = vz; }
    }

    auto stage = [&](int kt, int buf) {
        const int k0 = kt << 5;
        async_load_b128(&A[aOff0 + k0], &ldsA[buf][lA0]);
        async_load_b128(&A[aOff1 + k0], &ldsA[buf][lA1]);
        if (bOk0) async_load_b128(&Bt[bOff0 + k0], &ldsB[buf][lA0]);
        if (bOk1) async_load_b128(&Bt[bOff1 + k0], &ldsB[buf][lA1]);
    };

    stage(0, 0);
    for (int kt = 0; kt < kTiles; ++kt) {
        const int buf = kt & 1;
        wait_async0();            // own tile-kt transfers complete
        __syncthreads();          // everyone's transfers complete
        if (kt + 1 < kTiles) stage(kt + 1, buf ^ 1);  // overlap with WMMA below
        compute(&ldsA[buf][0], &ldsB[buf][0]);
        __syncthreads();          // done reading buf before it is refilled
    }
#else
    __shared__ __align__(32) bf16_t ldsA1[128 * 32];
    __shared__ __align__(32) bf16_t ldsB1[128 * 32];

    v8bf ra0, ra1, rb0, rb1;
    auto fetch = [&](int kt) {
        const int k0 = kt << 5;
        ra0 = *(const v8bf*)&A[aOff0 + k0];
        ra1 = *(const v8bf*)&A[aOff1 + k0];
        v8bf vz;
        #pragma unroll
        for (int i = 0; i < 8; ++i) vz[i] = bf16_zero();
        rb0 = vz; rb1 = vz;
        if (bOk0) rb0 = *(const v8bf*)&Bt[bOff0 + k0];
        if (bOk1) rb1 = *(const v8bf*)&Bt[bOff1 + k0];
    };

    fetch(0);
    for (int kt = 0; kt < kTiles; ++kt) {
        // commit staged registers (load-wait lands here, after a full WMMA phase)
        *(v8bf*)&ldsA1[lA0] = ra0;
        *(v8bf*)&ldsA1[lA1] = ra1;
        *(v8bf*)&ldsB1[lA0] = rb0;
        *(v8bf*)&ldsB1[lA1] = rb1;
        __syncthreads();
        if (kt + 1 < kTiles) fetch(kt + 1);   // issue next-tile loads, no wait yet
        compute(ldsA1, ldsB1);
        __syncthreads();
    }
#endif

    // epilogue: C VGPR r -> M = r + half*8 ; N = lane16
    #pragma unroll
    for (int j = 0; j < 8; ++j) {
        int n = n0 + j * 16 + lane16;
        if (n < Nreal) {
            float bv = bias ? bias[n] : 0.f;
            #pragma unroll
            for (int r = 0; r < 8; ++r) {
                int m = m0 + wave * 16 + half * 8 + r;
                float v = acc[j][r] + bv;
                if (useTanh) v = tanhf(v);
                C[(size_t)m * ldC + n] = v;
            }
        }
    }
}

// ---------------- leaf main layer: t = tanh(d[:,l,0:10] @ Wl_l + bl_l) ----------------
__global__ void leaf_kernel(const float* __restrict__ d, const float* __restrict__ Wl,
                            const float* __restrict__ bl, float* __restrict__ hl)
{
    int gid = blockIdx.x * 256 + threadIdx.x;
    if (gid >= NL * NB) return;
    int l = gid / NB, b = gid % NB;
    const float* drow = d + (size_t)b * N1 + l * NGP;
    float acc[NDL];
    const float* blp = bl + l * NDL;
    #pragma unroll
    for (int dl = 0; dl < NDL; ++dl) acc[dl] = blp[dl];
    #pragma unroll
    for (int g = 0; g < NG; ++g) {
        float dv = drow[g];
        const float* wrow = Wl + ((size_t)l * NG + g) * NDL;
        #pragma unroll
        for (int dl = 0; dl < NDL; ++dl) acc[dl] += dv * wrow[dl];
    }
    float* o = hl + (size_t)gid * NDL;
    #pragma unroll
    for (int dl = 0; dl < NDL; ++dl) o[dl] = tanhf(acc[dl]);
}

// ---------------- BatchNorm1d (training mode, biased var) in place ----------------
// one block per channel; h laid out [T][Bn][D], channel = blockIdx.x = t*D + d
__global__ __launch_bounds__(256) void bn_kernel(float* __restrict__ h,
    const float* __restrict__ gamma, const float* __restrict__ beta, int D, int Bn)
{
    const int ch = blockIdx.x;
    const int t = ch / D, dd = ch % D;
    float* base = h + (size_t)t * Bn * D + dd;
    const int tid = threadIdx.x;
    float s = 0.f, s2 = 0.f;
    for (int b = tid; b < Bn; b += 256) {
        float v = base[(size_t)b * D];
        s += v; s2 += v * v;
    }
    __shared__ float red[256], red2[256];
    red[tid] = s; red2[tid] = s2;
    __syncthreads();
    for (int off = 128; off > 0; off >>= 1) {
        if (tid < off) { red[tid] += red[tid + off]; red2[tid] += red2[tid + off]; }
        __syncthreads();
    }
    float mean  = red[0] / (float)Bn;
    float var   = red2[0] / (float)Bn - mean * mean;
    float scale = gamma[ch] * rsqrtf(var + 1e-5f);
    float shift = beta[ch] - mean * scale;
    for (int b = tid; b < Bn; b += 256)
        base[(size_t)b * D] = base[(size_t)b * D] * scale + shift;
}

// ---------------- aux head: out = tanh(h . Wa1 + ba1) * Wa2 + ba2 ----------------
__global__ void aux_kernel(const float* __restrict__ h, const float* __restrict__ Wa1,
    const float* __restrict__ ba1, const float* __restrict__ Wa2,
    const float* __restrict__ ba2, float* __restrict__ out, int D, int Bn, int total)
{
    int gid = blockIdx.x * 256 + threadIdx.x;
    if (gid >= total) return;
    int t = gid / Bn, b = gid % Bn;
    const float* row = h + ((size_t)t * Bn + b) * D;
    const float* w = Wa1 + (size_t)t * D;
    float s = ba1[t];
    for (int d = 0; d < D; ++d) s += row[d] * w[d];
    out[(size_t)t * Bn + b] = tanhf(s) * Wa2[t] + ba2[t];
}

// ---------------- host ----------------
extern "C" void kernel_launch(void* const* d_in, const int* in_sizes, int n_in,
                              void* d_out, int out_size, void* d_ws, size_t ws_size,
                              hipStream_t stream) {
    const float* x    = (const float*)d_in[0];
    const float* Wd   = (const float*)d_in[1];
    const float* bd   = (const float*)d_in[2];
    const float* Wl   = (const float*)d_in[3];
    const float* bl   = (const float*)d_in[4];
    const float* gl   = (const float*)d_in[5];
    const float* betal= (const float*)d_in[6];
    const float* Wa1l = (const float*)d_in[7];
    const float* ba1l = (const float*)d_in[8];
    const float* Wa2l = (const float*)d_in[9];
    const float* ba2l = (const float*)d_in[10];
    const float* Wm   = (const float*)d_in[11];
    const float* bm   = (const float*)d_in[12];
    const float* gm   = (const float*)d_in[13];
    const float* betam= (const float*)d_in[14];
    const float* Wa1m = (const float*)d_in[15];
    const float* ba1m = (const float*)d_in[16];
    const float* Wa2m = (const float*)d_in[17];
    const float* ba2m = (const float*)d_in[18];
    const float* Wr   = (const float*)d_in[19];
    const float* br   = (const float*)d_in[20];
    const float* gr   = (const float*)d_in[21];
    const float* betar= (const float*)d_in[22];
    const float* Wa1r = (const float*)d_in[23];
    const float* ba1r = (const float*)d_in[24];
    const float* Wa2r = (const float*)d_in[25];
    const float* ba2r = (const float*)d_in[26];

    float* out   = (float*)d_out;
    float* aux_l = out;
    float* aux_m = out + 786432;
    float* aux_r = out + 819200;
    float* h_l   = out + 821248;
    float* h_m   = out + 12617728;
    float* h_r   = out + 14977024;

    char* ws = (char*)d_ws;
    size_t off = 0;
    auto wsa = [&](size_t bytes) -> void* {
        void* p = ws + off;
        off = (off + bytes + 255) & ~(size_t)255;
        return p;
    };
    bf16_t* xb    = (bf16_t*)wsa((size_t)NB * NFP * 2);
    bf16_t* wdT   = (bf16_t*)wsa((size_t)N1 * NFP * 2);
    float*  biasd = (float*) wsa((size_t)N1 * 4);
    float*  dbuf  = (float*) wsa((size_t)NB * N1 * 4);
    bf16_t* cinb  = (bf16_t*)wsa((size_t)NM * NB * KMID * 2);
    bf16_t* wmT   = (bf16_t*)wsa((size_t)NM * NDMP * KMID * 2);
    bf16_t* rinb  = (bf16_t*)wsa((size_t)NB * KROOT * 2);
    bf16_t* wrT   = (bf16_t*)wsa((size_t)NDRP * KROOT * 2);

    // ---- stage 1: big leaf-input GEMM d = X @ WdT + bd ----
    cast_x_kernel   <<<(NB * NFP + 255) / 256, 256, 0, stream>>>(x, xb);
    cast_wd_kernel  <<<(N1 * NFP + 255) / 256, 256, 0, stream>>>(Wd, wdT);
    build_biasd_kernel<<<(N1 + 255) / 256, 256, 0, stream>>>(bd, biasd);
    wmma_gemm_kernel<<<dim3(NB / 128, N1 / 128, 1), 256, 0, stream>>>(
        xb, wdT, dbuf, biasd, NFP, N1, N1, N1, /*tanh=*/0, 0, 0, 0, 0);

    // ---- leaf main linear + tanh, then BN, then aux ----
    leaf_kernel<<<(NL * NB + 255) / 256, 256, 0, stream>>>(dbuf, Wl, bl, h_l);
    bn_kernel  <<<NL * NDL, 256, 0, stream>>>(h_l, gl, betal, NDL, NB);
    aux_kernel <<<(NL * NB + 255) / 256, 256, 0, stream>>>(
        h_l, Wa1l, ba1l, Wa2l, ba2l, aux_l, NDL, NB, NL * NB);

    // ---- mid: gather children, batched WMMA GEMM (+bias+tanh), BN, aux ----
    cast_cin_kernel<<<(NM * NB * KMID + 255) / 256, 256, 0, stream>>>(h_l, cinb);
    cast_wm_kernel <<<(NM * NDMP * KMID + 255) / 256, 256, 0, stream>>>(Wm, wmT);
    wmma_gemm_kernel<<<dim3(NB / 128, 1, NM), 256, 0, stream>>>(
        cinb, wmT, h_m, bm, KMID, NDMP, NDM, NDM, /*tanh=*/1,
        (long long)NB * KMID, (long long)NDMP * KMID, (long long)NB * NDM, NDM);
    bn_kernel <<<NM * NDM, 256, 0, stream>>>(h_m, gm, betam, NDM, NB);
    aux_kernel<<<(NM * NB + 255) / 256, 256, 0, stream>>>(
        h_m, Wa1m, ba1m, Wa2m, ba2m, aux_m, NDM, NB, NM * NB);

    // ---- root: gather mids, WMMA GEMM (+bias+tanh), BN, aux ----
    cast_rin_kernel<<<(NB * KROOT + 255) / 256, 256, 0, stream>>>(h_m, rinb);
    cast_wr_kernel <<<(NDRP * KROOT + 255) / 256, 256, 0, stream>>>(Wr, wrT);
    wmma_gemm_kernel<<<dim3(NB / 128, (NDRP + 127) / 128, 1), 256, 0, stream>>>(
        rinb, wrT, h_r, br, KROOT, NDRP, NDR, NDR, /*tanh=*/1, 0, 0, 0, 0);
    bn_kernel <<<NDR, 256, 0, stream>>>(h_r, gr, betar, NDR, NB);
    aux_kernel<<<(NB + 255) / 256, 256, 0, stream>>>(
        h_r, Wa1r, ba1r, Wa2r, ba2r, aux_r, NDR, NB, NB);
}
